// CrossAttention_3779571220602
// MI455X (gfx1250) — matrix operand order repaired and checked by reading, not compiled, not run on hardware
//
#include <hip/hip_runtime.h>
#include <hip/hip_bf16.h>
#include <math.h>

// ---------------------------------------------------------------------------
// CDNA5 (gfx1250) cross-attention: bf16 WMMA, f32 accumulate, wave32.
// - All WMMA operands load as contiguous 16B chunks (global_load_b128).
// - Weights and V pre-transposed so B-operands are K-fastest in memory.
// - K-loops software-pipelined (double-buffered) to overlap loads with WMMA.
// - GEMM uses 32x64 tiles: each B fragment feeds 2 WMMAs.
// ---------------------------------------------------------------------------

typedef __attribute__((ext_vector_type(16))) __bf16 v16bf;
typedef __attribute__((ext_vector_type(8)))  __bf16 v8bf;
typedef __attribute__((ext_vector_type(8)))  float  v8f;

#define WMMA_BF16(a, b, c) \
  __builtin_amdgcn_wmma_f32_16x16x32_bf16(false, (a), false, (b), (short)0, (c), false, false)

static __device__ __forceinline__ v8f vzero8() {
  v8f z;
#pragma unroll
  for (int i = 0; i < 8; ++i) z[i] = 0.0f;
  return z;
}

// A-matrix 16x32 bf16 fragment (contiguous K order, matching load_bT_frag). Lane L: row m = L&15.
// Slots 0..7  <-> K = (L>=16 ? 16 : 0) + 0..7   (contiguous 16B)
// Slots 8..15 <-> K = (L>=16 ? 16 : 0) + 8..15  (contiguous 16B)
static __device__ __forceinline__ v16bf load_a_frag(const __bf16* base, int ld, int lane) {
  const int m = lane & 15, kh = (lane >> 4) & 1;
  const __bf16* rp = base + (size_t)m * ld + kh * 16;
  union { v16bf v; v8bf h[2]; } u;
  u.h[0] = *(const v8bf*)(rp);
  u.h[1] = *(const v8bf*)(rp + 8);
  return u.v;
}

// B-matrix 32x16 bf16 fragment from K-fastest storage:
// element (k, n) lives at base[n*ld + k]. Lane L: n = L&15,
// slots s -> K = (L>=16 ? 16 : 0) + s  => 32 contiguous bytes per lane.
static __device__ __forceinline__ v16bf load_bT_frag(const __bf16* base, int ld, int lane) {
  const int n = lane & 15, kh = (lane >> 4) & 1;
  const __bf16* p = base + (size_t)n * ld + kh * 16;
  union { v16bf v; v8bf h[2]; } u;
  u.h[0] = *(const v8bf*)(p);
  u.h[1] = *(const v8bf*)(p + 8);
  return u.v;
}

// Reductions across the 16-lane half-wave sharing a row group (wave32).
static __device__ __forceinline__ float red_max16(float v) {
#pragma unroll
  for (int m = 1; m <= 8; m <<= 1) v = fmaxf(v, __shfl_xor(v, m, 32));
  return v;
}
static __device__ __forceinline__ float red_sum16(float v) {
#pragma unroll
  for (int m = 1; m <= 8; m <<= 1) v += __shfl_xor(v, m, 32);
  return v;
}

// ---------------------------------------------------------------------------
__global__ void cvt_f32_bf16(const float* __restrict__ in, __bf16* __restrict__ out, int n) {
  const int i = blockIdx.x * blockDim.x + threadIdx.x;
  if (i < n) out[i] = (__bf16)in[i];
}

// W (K x N rowmajor, f32) -> Wt (N x K rowmajor, bf16)
__global__ void cvt_transpose_bf16(const float* __restrict__ in, __bf16* __restrict__ out,
                                   int K, int N) {
  const int i = blockIdx.x * blockDim.x + threadIdx.x;
  if (i < K * N) {
    const int k = i / N, n = i % N;
    out[(size_t)n * K + k] = (__bf16)in[i];
  }
}

// ---------------------------------------------------------------------------
// GEMM: out(M x N) = A(M x K bf16 rowmajor) @ W, with W given TRANSPOSED
// (Bt is N x K rowmajor bf16). One wave per 32x64 tile (64 cols == one head).
// Software-pipelined K loop; each B fragment is reused by 2 WMMAs.
// mode 0: bf16 out rowmajor
// mode 1: per-head row L2 norm, bf16 out                (K projection)
// mode 2: +bias, L2 norm, * exp(min(slog[h], ln100))    (Q projection)
// mode 3: +bias, f32 out                                 (output projection)
// mode 4: bf16 out transposed per head: vt[((b*H+h)*Dh+d)*Ldim + l]  (V)
__global__ void gemm_wmma(const __bf16* __restrict__ A, const __bf16* __restrict__ Bt,
                          const float* __restrict__ bias, const float* __restrict__ slog,
                          __bf16* __restrict__ outb, float* __restrict__ outf,
                          int M, int N, int K, int mode, int Hh, int Ldim) {
  const int lane    = threadIdx.x;
  const int rowbase = blockIdx.x * 32;
  const int colbase = blockIdx.y * 64;

  v8f c[2][4];
#pragma unroll
  for (int mh = 0; mh < 2; ++mh)
#pragma unroll
    for (int t = 0; t < 4; ++t) c[mh][t] = vzero8();

  const __bf16* Ap0 = A + (size_t)rowbase * K;
  const __bf16* Ap1 = Ap0 + (size_t)16 * K;

  // prologue loads (k0 = 0)
  v16bf a0 = load_a_frag(Ap0, K, lane);
  v16bf a1 = load_a_frag(Ap1, K, lane);
  v16bf bb[4];
#pragma unroll
  for (int t = 0; t < 4; ++t)
    bb[t] = load_bT_frag(Bt + (size_t)(colbase + 16 * t) * K, K, lane);

  for (int k0 = 0; k0 < K; k0 += 32) {
    const int kn = (k0 + 32 < K) ? (k0 + 32) : 0;  // wrap keeps loads valid/uniform

    __builtin_prefetch(Ap0 + (size_t)(lane & 15) * K + k0 + 64, 0, 1);
    __builtin_prefetch(Ap1 + (size_t)(lane & 15) * K + k0 + 64, 0, 1);
    v16bf a0n = load_a_frag(Ap0 + kn, K, lane);
    v16bf a1n = load_a_frag(Ap1 + kn, K, lane);
    v16bf bn[4];
#pragma unroll
    for (int t = 0; t < 4; ++t) {
      const __bf16* bp = Bt + (size_t)(colbase + 16 * t) * K;
      __builtin_prefetch(bp + (size_t)(lane & 15) * K + k0 + 64, 0, 1);
      bn[t] = load_bT_frag(bp + kn, K, lane);
    }

#pragma unroll
    for (int t = 0; t < 4; ++t) {
      c[0][t] = WMMA_BF16(a0, bb[t], c[0][t]);
      c[1][t] = WMMA_BF16(a1, bb[t], c[1][t]);
    }
    a0 = a0n; a1 = a1n;
#pragma unroll
    for (int t = 0; t < 4; ++t) bb[t] = bn[t];
  }

  const int n = lane & 15, kh = (lane >> 4) & 1;

  if (bias != nullptr) {
#pragma unroll
    for (int t = 0; t < 4; ++t) {
      const float bv = bias[colbase + 16 * t + n];
#pragma unroll
      for (int mh = 0; mh < 2; ++mh)
#pragma unroll
        for (int r = 0; r < 8; ++r) c[mh][t][r] += bv;
    }
  }

  if (mode == 1 || mode == 2) {
    float sc = 1.0f;
    if (mode == 2) sc = __expf(fminf(slog[colbase >> 6], 4.6051701860f));  // ln(100)
#pragma unroll
    for (int mh = 0; mh < 2; ++mh) {
#pragma unroll
      for (int r = 0; r < 8; ++r) {
        float ss = 0.0f;
#pragma unroll
        for (int t = 0; t < 4; ++t) ss += c[mh][t][r] * c[mh][t][r];
        ss = red_sum16(ss);  // full 64-wide head row
        const float inv = sc / fmaxf(sqrtf(ss), 1e-12f);
#pragma unroll
        for (int t = 0; t < 4; ++t) c[mh][t][r] *= inv;
      }
    }
  }

  if (mode == 4) {
    const int h = colbase >> 6;  // Dh == 64
#pragma unroll
    for (int mh = 0; mh < 2; ++mh) {
#pragma unroll
      for (int t = 0; t < 4; ++t) {
#pragma unroll
        for (int r = 0; r < 8; ++r) {
          const int row = rowbase + 16 * mh + r + 8 * kh;  // = b*Ldim + l
          const int bb_ = row / Ldim, l = row % Ldim;
          const size_t idx = ((size_t)(bb_ * Hh + h) * 64 + 16 * t + n) * Ldim + l;
          outb[idx] = (__bf16)c[mh][t][r];
        }
      }
    }
    return;
  }

#pragma unroll
  for (int mh = 0; mh < 2; ++mh) {
#pragma unroll
    for (int t = 0; t < 4; ++t) {
#pragma unroll
      for (int r = 0; r < 8; ++r) {
        const size_t idx =
            (size_t)(rowbase + 16 * mh + r + 8 * kh) * N + colbase + 16 * t + n;
        if (mode == 3) outf[idx] = c[mh][t][r];
        else           outb[idx] = (__bf16)c[mh][t][r];
      }
    }
  }
}

// ---------------------------------------------------------------------------
// Flash attention. q/k: (B, L, H, D) rowmajor bf16 (normalized).
// vt: (B, H, D, Ly) bf16 (per-head transposed -> B-operand is K-fastest).
// One wave owns 16 query rows of one (b,h); streams 32 keys per step.
// K^T fragments are double-buffered across key-tile iterations.
__global__ void attn_wmma(const __bf16* __restrict__ qn, const __bf16* __restrict__ kn,
                          const __bf16* __restrict__ vt, const float* __restrict__ bias,
                          __bf16* __restrict__ out,
                          int Bb, int Lq, int Ly, int H, int Dh) {
  const int C    = H * Dh;
  const int lane = threadIdx.x;
  const int qtiles = Lq / 16;
  int bid = blockIdx.x;
  const int qt = bid % qtiles; bid /= qtiles;
  const int h  = bid % H;
  const int b  = bid / H;
  const int qbase = qt * 16;
  const int n = lane & 15, kh = (lane >> 4) & 1;

  const __bf16* qp = qn + ((size_t)(b * Lq + qbase)) * C + h * Dh;
  const v16bf aq0 = load_a_frag(qp,      C, lane);  // dims 0..31
  const v16bf aq1 = load_a_frag(qp + 32, C, lane);  // dims 32..63

  const __bf16* kbase = kn + ((size_t)b * Ly) * C + h * Dh;
  const __bf16* vbase = vt + ((size_t)(b * H + h) * Dh) * Ly;

  float mrow[8], lrow[8];
  v8f o[4];
#pragma unroll
  for (int r = 0; r < 8; ++r) { mrow[r] = -1e30f; lrow[r] = 0.0f; }
#pragma unroll
  for (int t = 0; t < 4; ++t) o[t] = vzero8();

  __shared__ __align__(16) __bf16 pshm[16 * 32];  // P tile: C-layout -> A-layout

  // prologue: K^T fragments for first key tile.
  // bk[2*t + ch] = keys (16t..16t+15), head dims (32ch..32ch+31)
  v16bf bk[4];
#pragma unroll
  for (int t = 0; t < 2; ++t)
#pragma unroll
    for (int ch = 0; ch < 2; ++ch)
      bk[2 * t + ch] = load_bT_frag(kbase + (size_t)(16 * t) * C + 32 * ch, C, lane);

  for (int j0 = 0; j0 < Ly; j0 += 32) {
    const int jn = (j0 + 32 < Ly) ? (j0 + 32) : 0;

    // S = Q @ K^T from current buffers
    v8f s[2];
#pragma unroll
    for (int t = 0; t < 2; ++t) {
      v8f z = vzero8();
      z = WMMA_BF16(aq0, bk[2 * t + 0], z);
      z = WMMA_BF16(aq1, bk[2 * t + 1], z);
      s[t] = z;
    }

    // issue next key-tile loads + prefetch V; softmax VALU below overlaps them
    v16bf bkn[4];
#pragma unroll
    for (int t = 0; t < 2; ++t)
#pragma unroll
      for (int ch = 0; ch < 2; ++ch)
        bkn[2 * t + ch] =
            load_bT_frag(kbase + (size_t)(jn + 16 * t) * C + 32 * ch, C, lane);
    __builtin_prefetch(vbase + (size_t)(lane & 15) * Ly + j0, 0, 1);
    __builtin_prefetch(vbase + (size_t)(32 + (lane & 15)) * Ly + j0, 0, 1);

    // + attn_bias (Lq x Ly)
#pragma unroll
    for (int t = 0; t < 2; ++t)
#pragma unroll
      for (int r = 0; r < 8; ++r)
        s[t][r] += bias[(size_t)(qbase + r + 8 * kh) * Ly + j0 + 16 * t + n];

    // online softmax
    float corr[8];
#pragma unroll
    for (int r = 0; r < 8; ++r) {
      const float tm = red_max16(fmaxf(s[0][r], s[1][r]));
      const float mn = fmaxf(mrow[r], tm);
      corr[r] = __expf(mrow[r] - mn);   // exp(-huge) = 0 on first tile
      mrow[r] = mn;
      const float p0 = __expf(s[0][r] - mn);
      const float p1 = __expf(s[1][r] - mn);
      s[0][r] = p0; s[1][r] = p1;
      lrow[r] = lrow[r] * corr[r] + red_sum16(p0 + p1);
    }
#pragma unroll
    for (int t = 0; t < 4; ++t)
#pragma unroll
      for (int r = 0; r < 8; ++r) o[t][r] *= corr[r];

    // P: C-fragment -> A-fragment via LDS (re-read is 2x ds_load_b128/lane)
#pragma unroll
    for (int t = 0; t < 2; ++t)
#pragma unroll
      for (int r = 0; r < 8; ++r)
        pshm[(r + 8 * kh) * 32 + 16 * t + n] = (__bf16)s[t][r];
    __syncthreads();
    const v16bf ap = load_a_frag(pshm, 32, lane);

    // O += P @ V   (V pre-transposed: key index is fastest)
#pragma unroll
    for (int d = 0; d < 4; ++d) {
      const v16bf bv = load_bT_frag(vbase + (size_t)(16 * d) * Ly + j0, Ly, lane);
      o[d] = WMMA_BF16(ap, bv, o[d]);
    }
    __syncthreads();

#pragma unroll
    for (int i = 0; i < 4; ++i) bk[i] = bkn[i];
  }

  // normalize, write (B, Lq, H, D) bf16
#pragma unroll
  for (int r = 0; r < 8; ++r) {
    const float inv = 1.0f / lrow[r];
#pragma unroll
    for (int d = 0; d < 4; ++d) {
      const size_t idx = ((size_t)(b * Lq + qbase + r + 8 * kh)) * C + h * Dh + 16 * d + n;
      out[idx] = (__bf16)(o[d][r] * inv);
    }
  }
}

// ---------------------------------------------------------------------------
extern "C" void kernel_launch(void* const* d_in, const int* in_sizes, int n_in,
                              void* d_out, int out_size, void* d_ws, size_t ws_size,
                              hipStream_t stream) {
  const int Bb = 4, Lq = 1024, Ly = 2048, C = 1024, H = 16, Dh = 64;
  (void)in_sizes; (void)n_in; (void)out_size; (void)ws_size;

  const float* x    = (const float*)d_in[0];
  const float* y    = (const float*)d_in[1];
  const float* abia = (const float*)d_in[2];
  const float* Wq   = (const float*)d_in[3];
  const float* bq   = (const float*)d_in[4];
  const float* Wk   = (const float*)d_in[5];
  const float* Wv   = (const float*)d_in[6];
  const float* Wo   = (const float*)d_in[7];
  const float* bo   = (const float*)d_in[8];
  const float* slog = (const float*)d_in[9];
  float* out = (float*)d_out;

  const size_t nx = (size_t)Bb * Lq * C;  // 4,194,304
  const size_t ny = (size_t)Bb * Ly * C;  // 8,388,608
  const size_t nw = (size_t)C * C;        // 1,048,576

  __bf16* p   = (__bf16*)d_ws;
  __bf16* xb  = p; p += nx;
  __bf16* yb  = p; p += ny;
  __bf16* wqt = p; p += nw;   // W^T (N x K) bf16
  __bf16* wkt = p; p += nw;
  __bf16* wvt = p; p += nw;
  __bf16* wot = p; p += nw;
  __bf16* qnb = p; p += nx;   // normalized+scaled Q, (B,Lq,H,D)
  __bf16* knb = p; p += ny;   // normalized K, (B,Ly,H,D)
  __bf16* vtb = p; p += ny;   // V transposed, (B,H,D,Ly)
  __bf16* aob = p; p += nx;   // attention output, (B,Lq,H,D)

  auto cvt = [&](const float* s, __bf16* d, size_t nel) {
    cvt_f32_bf16<<<dim3((unsigned)((nel + 255) / 256)), dim3(256), 0, stream>>>(s, d, (int)nel);
  };
  auto cvtT = [&](const float* s, __bf16* d) {
    cvt_transpose_bf16<<<dim3((unsigned)((nw + 255) / 256)), dim3(256), 0, stream>>>(s, d, C, C);
  };
  cvt(x, xb, nx);  cvt(y, yb, ny);
  cvtT(Wq, wqt); cvtT(Wk, wkt); cvtT(Wv, wvt); cvtT(Wo, wot);

  // Q = l2norm(x@Wq + bq) * exp(min(slog, ln100))
  gemm_wmma<<<dim3((Bb * Lq) / 32, C / 64), dim3(32), 0, stream>>>(
      xb, wqt, bq, slog, qnb, nullptr, Bb * Lq, C, C, 2, H, Ly);
  // K = l2norm(y@Wk)
  gemm_wmma<<<dim3((Bb * Ly) / 32, C / 64), dim3(32), 0, stream>>>(
      yb, wkt, nullptr, nullptr, knb, nullptr, Bb * Ly, C, C, 1, H, Ly);
  // V = y@Wv, written per-head transposed (B,H,D,Ly)
  gemm_wmma<<<dim3((Bb * Ly) / 32, C / 64), dim3(32), 0, stream>>>(
      yb, wvt, nullptr, nullptr, vtb, nullptr, Bb * Ly, C, C, 4, H, Ly);
  // softmax(QK^T + bias) @ V
  attn_wmma<<<dim3(Bb * H * (Lq / 16)), dim3(32), 0, stream>>>(
      qnb, knb, vtb, abia, aob, Bb, Lq, Ly, H, Dh);
  // out = AO @ Wo + bo   (f32)
  gemm_wmma<<<dim3((Bb * Lq) / 32, C / 64), dim3(32), 0, stream>>>(
      aob, wot, bo, nullptr, nullptr, out, Bb * Lq, C, C, 3, H, Ly);
}
